// TreeLSTM_38371237822587
// MI455X (gfx1250) — matrix operand order repaired and checked by reading, not compiled, not run on hardware
//
#include <hip/hip_runtime.h>
#include <hip/hip_bf16.h>

typedef __attribute__((ext_vector_type(16))) _Float16 v16h;
typedef __attribute__((ext_vector_type(8)))  _Float16 v8h;
typedef __attribute__((ext_vector_type(8)))  float    v8f;

#define HDIM      256
#define HDIM2     512
#define N_LEAVES  2048
#define N_INTERNAL 2047
#define N_NODES   4095
#define NUM_CLASSES 5

// Workspace layout (offsets in _Float16 elements; all 16B-aligned)
#define OFF_H    0                         // h for all nodes, padded to 4096 rows
#define OFF_E    (4096 * 256)              // gathered leaf embeddings (f16)
#define OFF_WI   (OFF_E  + 2048 * 256)
#define OFF_WO   (OFF_WI + 256 * 256)
#define OFF_WU   (OFF_WO + 256 * 256)
#define OFF_UI   (OFF_WU + 256 * 256)
#define OFF_UO   (OFF_UI + 256 * 512)
#define OFF_UU   (OFF_UO + 256 * 512)
#define OFF_UF1  (OFF_UU + 256 * 512)
#define OFF_UF2  (OFF_UF1 + 256 * 256)

__device__ __forceinline__ float sigf(float x) { return 1.0f / (1.0f + __expf(-x)); }

// A-tile (16x32 f16) per CDNA5 layout: lane L holds row L&15;
// half j -> K = (j>>3)*16 + (L>>4)*8 + (j&7). rowPtr points at k_base of this row.
__device__ __forceinline__ v16h loadA(const _Float16* rowPtr, int lane) {
    int koff = (lane >> 4) * 8;
    v8h lo = *reinterpret_cast<const v8h*>(rowPtr + koff);
    v8h hi = *reinterpret_cast<const v8h*>(rowPtr + 16 + koff);
    v16h a;
#pragma unroll
    for (int j = 0; j < 8; ++j) { a[j] = lo[j]; a[j + 8] = hi[j]; }
    return a;
}

// B-tile (32x16 f16): lane L holds col L&15; half j -> K = (j>>3)*16 + (L>>4)*8 + (j&7).
// Since out = x @ W^T, B[k][n] = W[n][k]: colPtr = W + n*stride + k_base -> contiguous.
__device__ __forceinline__ v16h loadB(const _Float16* colPtr, int lane) {
    int koff = (lane >> 4) * 8;
    v8h lo = *reinterpret_cast<const v8h*>(colPtr + koff);
    v8h hi = *reinterpret_cast<const v8h*>(colPtr + 16 + koff);
    v16h b;
#pragma unroll
    for (int j = 0; j < 8; ++j) { b[j] = lo[j]; b[j + 8] = hi[j]; }
    return b;
}

#define WMMA(acc, a, b) \
    acc = __builtin_amdgcn_wmma_f32_16x16x32_f16(false, (a), false, (b), (short)0, (acc), false, false)

// ---------------- Prep: gather embeddings + convert all weights to f16 ----------------
__global__ __launch_bounds__(256) void tlstm_prep(
    const int* __restrict__ word_ids, const float* __restrict__ emb,
    const float* __restrict__ Wi, const float* __restrict__ Wo, const float* __restrict__ Wu,
    const float* __restrict__ Ui, const float* __restrict__ Uo, const float* __restrict__ Uu,
    const float* __restrict__ Uf1, const float* __restrict__ Uf2,
    _Float16* __restrict__ ws)
{
    int tid = blockIdx.x * blockDim.x + threadIdx.x;
    int stride = gridDim.x * blockDim.x;
    for (int i = tid; i < N_LEAVES * HDIM; i += stride) {
        int leaf = i >> 8, k = i & 255;
        ws[OFF_E + i] = (_Float16)emb[(long)word_ids[leaf] * HDIM + k];
    }
    for (int i = tid; i < HDIM * HDIM; i += stride) {
        ws[OFF_WI + i]  = (_Float16)Wi[i];
        ws[OFF_WO + i]  = (_Float16)Wo[i];
        ws[OFF_WU + i]  = (_Float16)Wu[i];
        ws[OFF_UF1 + i] = (_Float16)Uf1[i];
        ws[OFF_UF2 + i] = (_Float16)Uf2[i];
    }
    for (int i = tid; i < HDIM * HDIM2; i += stride) {
        ws[OFF_UI + i] = (_Float16)Ui[i];
        ws[OFF_UO + i] = (_Float16)Uo[i];
        ws[OFF_UU + i] = (_Float16)Uu[i];
    }
}

// ---------------- Leaf: 3 fused GEMMs [2048x256]x[256x256]^T + gates ----------------
// One wave per 16x16 output tile: 2048 waves -> 256 blocks of 256 threads.
__global__ __launch_bounds__(256) void tlstm_leaf(
    const float* __restrict__ bi, const float* __restrict__ bo, const float* __restrict__ bu,
    _Float16* __restrict__ ws)
{
    const _Float16* E  = ws + OFF_E;
    const _Float16* WI = ws + OFF_WI;
    const _Float16* WO = ws + OFF_WO;
    const _Float16* WU = ws + OFF_WU;
    _Float16* H = ws + OFF_H;

    int lane = threadIdx.x & 31;
    int wave = (blockIdx.x * blockDim.x + threadIdx.x) >> 5;   // 0..2047
    int mBase = (wave >> 4) * 16;
    int nBase = (wave & 15) * 16;
    int n = nBase + (lane & 15);

    const _Float16* aRow = E + (mBase + (lane & 15)) * HDIM;
    const _Float16* bI = WI + n * HDIM;
    const _Float16* bO = WO + n * HDIM;
    const _Float16* bU = WU + n * HDIM;

    v8f ai = {}, ao = {}, au = {};
#pragma unroll
    for (int kt = 0; kt < 8; ++kt) {
        int kb = kt * 32;
        v16h a = loadA(aRow + kb, lane);
        WMMA(ai, a, loadB(bI + kb, lane));
        WMMA(ao, a, loadB(bO + kb, lane));
        WMMA(au, a, loadB(bU + kb, lane));
    }

    float bin = bi[n], bon = bo[n], bun = bu[n];
#pragma unroll
    for (int r = 0; r < 8; ++r) {
        int m = mBase + ((lane >> 4) * 8 + r);
        float iv = sigf(ai[r] + bin);
        float ov = sigf(ao[r] + bon);
        float uv = fmaxf(au[r] + bun, 0.0f);
        float c  = iv * uv;
        H[m * HDIM + n] = (_Float16)(ov * fmaxf(c, 0.0f));
    }
}

// ---------------- One tree level: P independent nodes, 5 fused GEMMs + gates ----------------
__global__ __launch_bounds__(256) void tlstm_level(
    const int* __restrict__ left, const int* __restrict__ right,
    const float* __restrict__ bui, const float* __restrict__ buo, const float* __restrict__ buu,
    const float* __restrict__ bf1, const float* __restrict__ bf2,
    _Float16* __restrict__ ws, int ordBase, int count)
{
    int lane = threadIdx.x & 31;
    int wave = (blockIdx.x * blockDim.x + threadIdx.x) >> 5;
    int rowTiles = (count + 15) >> 4;
    if (wave >= rowTiles * 16) return;          // wave-uniform exit (EXEC stays all-1 for WMMA)

    int mBase = (wave >> 4) * 16;
    int nBase = (wave & 15) * 16;
    int n = nBase + (lane & 15);

    const _Float16* H   = ws + OFF_H;
    const _Float16* UI  = ws + OFF_UI;
    const _Float16* UO  = ws + OFF_UO;
    const _Float16* UU  = ws + OFF_UU;
    const _Float16* UF1 = ws + OFF_UF1;
    const _Float16* UF2 = ws + OFF_UF2;

    // A-row gather: row m = concat(h[left], h[right]); clamp OOB rows to a valid node.
    int rowA = mBase + (lane & 15);
    int ordA = ordBase + ((rowA < count) ? rowA : (count - 1));
    const _Float16* hl = H + left[ordA]  * HDIM;
    const _Float16* hr = H + right[ordA] * HDIM;

    v8f accI = {}, accO = {}, accU = {}, accF1 = {}, accF2 = {};
#pragma unroll
    for (int kt = 0; kt < 16; ++kt) {
        int kb = kt * 32;
        const _Float16* src = (kb < 256) ? (hl + kb) : (hr + (kb - 256));
        v16h a = loadA(src, lane);
        WMMA(accI, a, loadB(UI + n * HDIM2 + kb, lane));
        WMMA(accO, a, loadB(UO + n * HDIM2 + kb, lane));
        WMMA(accU, a, loadB(UU + n * HDIM2 + kb, lane));
        if (kt < 8) { WMMA(accF1, a, loadB(UF1 + n * HDIM + kb, lane)); }
        else        { WMMA(accF2, a, loadB(UF2 + n * HDIM + (kb - 256), lane)); }
    }

    float bI = bui[n], bO = buo[n], bU = buu[n], b1 = bf1[n], b2 = bf2[n];
    _Float16* Hw = ws + OFF_H;
#pragma unroll
    for (int r = 0; r < 8; ++r) {
        int row = mBase + ((lane >> 4) * 8 + r);
        if (row < count) {
            int ord  = ordBase + row;
            int node = N_LEAVES + ord;
            float lhv = (float)H[left[ord]  * HDIM + n];
            float rhv = (float)H[right[ord] * HDIM + n];
            float ii = sigf(accI[r] + bI);
            float oo = sigf(accO[r] + bO);
            float uu = fmaxf(accU[r] + bU, 0.0f);
            float f1 = sigf(accF1[r] + b1);
            float f2 = sigf(accF2[r] + b2);
            float c  = ii * uu + f1 * lhv + f2 * rhv;
            Hw[node * HDIM + n] = (_Float16)(oo * fmaxf(c, 0.0f));
        }
    }
}

// ---------------- Projection: logits[node][c] = h[node] . Wp[c] + bp[c] ----------------
__global__ __launch_bounds__(256) void tlstm_proj(
    const float* __restrict__ Wp, const float* __restrict__ bp,
    const _Float16* __restrict__ ws, float* __restrict__ out)
{
    int lane = threadIdx.x & 31;
    int wave = (blockIdx.x * blockDim.x + threadIdx.x) >> 5;
    if (wave >= N_NODES) return;
    const _Float16* h = ws + OFF_H + wave * HDIM;
    float acc[NUM_CLASSES] = {0.f, 0.f, 0.f, 0.f, 0.f};
    for (int k = lane; k < HDIM; k += 32) {
        float hv = (float)h[k];
#pragma unroll
        for (int c = 0; c < NUM_CLASSES; ++c) acc[c] += hv * Wp[c * HDIM + k];
    }
#pragma unroll
    for (int c = 0; c < NUM_CLASSES; ++c)
        for (int off = 16; off > 0; off >>= 1) acc[c] += __shfl_down(acc[c], off, 32);
    if (lane == 0) {
#pragma unroll
        for (int c = 0; c < NUM_CLASSES; ++c) out[wave * NUM_CLASSES + c] = acc[c] + bp[c];
    }
}

extern "C" void kernel_launch(void* const* d_in, const int* in_sizes, int n_in,
                              void* d_out, int out_size, void* d_ws, size_t ws_size,
                              hipStream_t stream)
{
    const int*   word_ids = (const int*)d_in[0];
    const int*   left     = (const int*)d_in[1];
    const int*   right    = (const int*)d_in[2];
    const float* emb      = (const float*)d_in[3];
    const float* Wi  = (const float*)d_in[4];   const float* bi  = (const float*)d_in[5];
    const float* Wo  = (const float*)d_in[6];   const float* bo  = (const float*)d_in[7];
    const float* Wu  = (const float*)d_in[8];   const float* bu  = (const float*)d_in[9];
    const float* Ui  = (const float*)d_in[10];  const float* bui = (const float*)d_in[11];
    const float* Uo  = (const float*)d_in[12];  const float* buo = (const float*)d_in[13];
    const float* Uu  = (const float*)d_in[14];  const float* buu = (const float*)d_in[15];
    const float* Uf1 = (const float*)d_in[16];  const float* bf1 = (const float*)d_in[17];
    const float* Uf2 = (const float*)d_in[18];  const float* bf2 = (const float*)d_in[19];
    const float* Wp  = (const float*)d_in[20];  const float* bp  = (const float*)d_in[21];

    _Float16* ws  = (_Float16*)d_ws;   // needs ~4.6 MB
    float*    out = (float*)d_out;

    tlstm_prep<<<1024, 256, 0, stream>>>(word_ids, emb, Wi, Wo, Wu, Ui, Uo, Uu, Uf1, Uf2, ws);

    tlstm_leaf<<<256, 256, 0, stream>>>(bi, bo, bu, ws);   // 2048 tiles, 1 wave each

    // Balanced tree: levels of 1024, 512, ..., 1 internal nodes, in topo order.
    int ordBase = 0;
    for (int cnt = N_LEAVES / 2; cnt >= 1; cnt >>= 1) {
        int rowTiles = (cnt + 15) >> 4;
        tlstm_level<<<rowTiles * 2, 256, 0, stream>>>(left, right, bui, buo, buu, bf1, bf2,
                                                      ws, ordBase, cnt);
        ordBase += cnt;
    }

    tlstm_proj<<<512, 256, 0, stream>>>(Wp, bp, ws, out);
}
